// RobustStateSpaceSimulatorAugmented_Non_normalized_120259084396
// MI455X (gfx1250) — matrix operand order repaired and checked by reading, not compiled, not run on hardware
//
#include <hip/hip_runtime.h>
#include <hip/hip_bf16.h>
#include <math.h>

typedef _Float16 half_t;
typedef __attribute__((ext_vector_type(16))) _Float16 v16h;
typedef __attribute__((ext_vector_type(8)))  float    v8f;

#define BSZ 512
#define TSZ 512
#define HD  512
#define NX  32
#define NXB 32
#define NU  16
#define NY  8
#define NTH 8
#define ZS  96          // LDS z-buffer stride (max padded din)
#define NTHREADS 256

// ---- workspace layout (in halfs) ------------------------------------------
// W1 panels: Kc * 32 Ntiles * 32 lanes * 16 halfs
#define OFF_W1_F   0        // Kc=2
#define OFF_W1_FB  32768    // Kc=3
#define OFF_W1_GB  81920    // Kc=3
#define OFF_W1_H   131072   // Kc=2
#define OFF_W1_HB  163840   // Kc=2
// W2 panels: 16 Kchunks * Nt2 * 32 lanes * 16 halfs
#define OFF_W2_F   196608   // Nt2=2
#define OFF_W2_FB  212992   // Nt2=2
#define OFF_W2_GB  229376   // Nt2=2
#define OFF_W2_H   245760   // Nt2=1
#define OFF_W2_HB  253952   // Nt2=1

// ---- output layout (floats) ------------------------------------------------
#define OUT_XF 0
#define OUT_XB (BSZ*TSZ*NX)
#define OUT_Y  (2*BSZ*TSZ*NX)
#define OUT_XP (2*BSZ*TSZ*NX + BSZ*TSZ*NY)

// ============================================================================
// Weight packing: fp32 -> fp16 in WMMA B-operand layout.
// Layout per 32x16 (KxN) chunk: lane<16 holds N=nt*16+lane, K_local=j (j<8), 16+j-8 (j>=8);
// lane>=16 holds N=nt*16+lane-16, K_local=8+j (j<8), 24+j-8 (j>=8).  16 contiguous halfs per lane.
// ============================================================================
__global__ void pack_w1(const float* __restrict__ W1, int din,
                        half_t* __restrict__ panel, int Kc) {
    int idx = blockIdx.x * blockDim.x + threadIdx.x;
    int total = Kc * 32 * 32 * 16;
    if (idx >= total) return;
    int j    = idx & 15;
    int lane = (idx >> 4) & 31;
    int nt   = (idx >> 9) & 31;
    int kc   = idx >> 14;
    int n = nt * 16 + (lane & 15);
    int K = kc * 32 + ((j & 8) << 1) + ((lane >> 4) << 3) + (j & 7);
    float v = (K < din) ? W1[K * HD + n] : 0.0f;
    panel[idx] = (half_t)v;
}

__global__ void pack_w2(const float* __restrict__ W2, int dout,
                        half_t* __restrict__ panel, int Nt2) {
    int idx = blockIdx.x * blockDim.x + threadIdx.x;
    int total = 16 * Nt2 * 32 * 16;
    if (idx >= total) return;
    int j    = idx & 15;
    int lane = (idx >> 4) & 31;
    int rest = idx >> 9;
    int nt   = rest % Nt2;
    int kc   = rest / Nt2;
    int n = nt * 16 + (lane & 15);
    int K = kc * 32 + ((j & 8) << 1) + ((lane >> 4) << 3) + (j & 7);
    float v = (n < dout) ? W2[K * dout + n] : 0.0f;
    panel[idx] = (half_t)v;
}

// ============================================================================
// Device helpers
// ============================================================================
__device__ __forceinline__ v16h bfrag(const half_t* p) {
    return *(const v16h*)p;   // 32B contiguous per lane -> 2x global_load_b128
}

// Gather A-operand fragment from row-major LDS matrix (16 rows x stride halfs).
// ISA 16-bit A 16x32 layout; contiguous K-octets merge into 2x ds_load_b128.
__device__ __forceinline__ v16h afrag(const half_t* z, int stride, int kc, int lane) {
    union { unsigned int u[8]; v16h h; } r;
    int m  = lane & 15;
    int hi = (lane & 16) >> 1;            // 0 or 8
    const half_t* row = z + m * stride + kc * 32;
#pragma unroll
    for (int v = 0; v < 8; ++v) {
        int K = ((v & 4) << 2) + hi + ((v & 3) << 1);  // paired K, 4B aligned
        r.u[v] = *(const unsigned int*)(row + K);
    }
    return r.h;
}

__device__ __forceinline__ float clipf(float v, float lo, float hi) {
    return fminf(fmaxf(v, lo), hi);
}

// Branch-free tanh. gfx1250 has v_tanh_f32; use it when the toolchain exposes
// the builtin, else ~7 VALU ops via native exp/rcp (no EXEC divergence).
__device__ __forceinline__ float fast_tanh(float x) {
#if __has_builtin(__builtin_amdgcn_tanhf)
    return __builtin_amdgcn_tanhf(x);
#else
    float ax = __builtin_fabsf(x);
    float t  = __expf(-2.0f * ax);                       // v_exp_f32 path
    float r  = (1.0f - t) * __builtin_amdgcn_rcpf(1.0f + t);
    return __builtin_copysignf(r, x);
#endif
}

__device__ __forceinline__ v8f wmma16(v16h a, v16h b, v8f c) {
    return __builtin_amdgcn_wmma_f32_16x16x32_f16(false, a, false, b,
                                                  (short)0, c, false, false);
}

// One MLP on a 16-row batch tile: out(16 x NT2*16) = tanh(z@W1+b1)@W2+b2
// Block = 256 threads (8 waves). GEMM1: wave w -> N-tiles 4w..4w+3 of hidden.
// GEMM2: wave w -> K-chunks 2w,2w+1; partials reduced via ds_add_f32.
// K-chunks go to alternating accumulators so WMMAs pipeline (no RAW NOPs).
template<int KC, int NT2>
__device__ __forceinline__ void mlp_tile(
    const half_t* __restrict__ z, int zstride,
    const half_t* __restrict__ W1p, const float* __restrict__ b1,
    const half_t* __restrict__ W2p, const float* __restrict__ b2,
    int dout,
    half_t* __restrict__ hid, float* __restrict__ acc,
    bool accumulate, int tid)
{
    const int lane  = tid & 31;
    const int wave  = tid >> 5;
    const int mbase = (lane & 16) >> 1;   // rows 0..7 or 8..15
    __syncthreads();                       // z ready, previous acc readers done

    // ---- GEMM1: hidden = tanh(z @ W1 + b1), written fp16 to LDS ----
#pragma unroll
    for (int i = 0; i < 4; ++i) {
        const int nt = wave * 4 + i;
        v8f c0 = {}, c1 = {};
#pragma unroll
        for (int kc = 0; kc < KC; ++kc) {
            v16h a = afrag(z, zstride, kc, lane);
            v16h b = bfrag(W1p + (((kc * 32 + nt) * 32 + lane) << 4));
            if (kc & 1) c1 = wmma16(a, b, c1);
            else        c0 = wmma16(a, b, c0);
        }
        const int n = nt * 16 + (lane & 15);
        const float bias = b1[n];
#pragma unroll
        for (int r = 0; r < 8; ++r) {
            float v = (KC > 1) ? (c0[r] + c1[r]) : c0[r];
            hid[(mbase + r) * HD + n] = (half_t)fast_tanh(v + bias);
        }
    }

    // ---- init (or bias-augment) output accumulator ----
    const int osz = 16 * NT2 * 16;
    for (int idx = tid; idx < osz; idx += NTHREADS) {
        const int col = idx % (NT2 * 16);
        const float bb = (col < dout) ? b2[col] : 0.0f;
        acc[idx] = accumulate ? (acc[idx] + bb) : bb;
    }
    __syncthreads();                       // hid + acc ready

    // ---- GEMM2: out += hid @ W2 (K split across waves) ----
#pragma unroll
    for (int nt = 0; nt < NT2; ++nt) {
        v8f c0 = {}, c1 = {};
        {
            const int kc0 = wave * 2;
            v16h a0 = afrag(hid, HD, kc0, lane);
            v16h b0 = bfrag(W2p + (((kc0 * NT2 + nt) * 32 + lane) << 4));
            c0 = wmma16(a0, b0, c0);
            const int kc1 = kc0 + 1;
            v16h a1 = afrag(hid, HD, kc1, lane);
            v16h b1v = bfrag(W2p + (((kc1 * NT2 + nt) * 32 + lane) << 4));
            c1 = wmma16(a1, b1v, c1);
        }
        const int ncol = nt * 16 + (lane & 15);
#pragma unroll
        for (int r = 0; r < 8; ++r)
            atomicAdd(&acc[(mbase + r) * (NT2 * 16) + ncol], c0[r] + c1[r]);
    }
    __syncthreads();
}

// ============================================================================
// Persistent simulation kernel: 32 blocks x 16 batch rows, full T scan.
// ============================================================================
__global__ __launch_bounds__(NTHREADS) void sim_kernel(
    const float* __restrict__ x_f_0, const float* __restrict__ x_b_0,
    const float* __restrict__ u,
    const float* __restrict__ theta_f_g, const float* __restrict__ theta_h_g,
    const float* __restrict__ u_mean, const float* __restrict__ u_std,
    const float* __restrict__ x_mean, const float* __restrict__ x_std,
    const float* __restrict__ y_mean, const float* __restrict__ y_std,
    const float* __restrict__ min_x_b, const float* __restrict__ max_x_b,
    const float* __restrict__ min_x, const float* __restrict__ max_x,
    const float* __restrict__ min_y, const float* __restrict__ max_y,
    const float* __restrict__ b1_f,  const float* __restrict__ b2_f,
    const float* __restrict__ b1_fb, const float* __restrict__ b2_fb,
    const float* __restrict__ b1_gb, const float* __restrict__ b2_gb,
    const float* __restrict__ b1_h,  const float* __restrict__ b2_h,
    const float* __restrict__ b1_hb, const float* __restrict__ b2_hb,
    const half_t* __restrict__ ws, float* __restrict__ out)
{
    __shared__ half_t s_z[16 * ZS];
    __shared__ half_t s_hid[16 * HD];
    __shared__ float s_acc_gb[16 * 32], s_acc_fb[16 * 32], s_acc_f[16 * 32];
    __shared__ float s_acc_y[16 * 16];
    __shared__ float s_xf[16 * 32], s_xp[16 * 32], s_xb[16 * 32], s_y[16 * 8];
    __shared__ float s_thf[16 * 8], s_thh[16 * 8], s_ut[16 * 16];
    __shared__ float c_xmean[32], c_xstd[32], c_xstdi[32];
    __shared__ float c_minxd[32], c_maxxd[32], c_minxb[32], c_maxxb[32];
    __shared__ float c_umean[16], c_ustd[16];
    __shared__ float c_ymean[8], c_ystdi[8], c_minyd[8], c_maxyd[8];

    const int tid   = threadIdx.x;
    const int rbase = blockIdx.x * 16;

    // ---- constants + denormalized clamp bounds ----
    if (tid < 32) {
        float xm = x_mean[tid], xs = x_std[tid];
        c_xmean[tid] = xm; c_xstd[tid] = xs; c_xstdi[tid] = 1.0f / xs;
        c_minxd[tid] = min_x[tid] * xs + xm;
        c_maxxd[tid] = max_x[tid] * xs + xm;
        c_minxb[tid] = min_x_b[tid];
        c_maxxb[tid] = max_x_b[tid];
    }
    if (tid < 16) { c_umean[tid] = u_mean[tid]; c_ustd[tid] = u_std[tid]; }
    if (tid < 8) {
        float ym = y_mean[tid], ysd = y_std[tid];
        c_ymean[tid] = ym; c_ystdi[tid] = 1.0f / ysd;
        c_minyd[tid] = min_y[tid] * ysd + ym;
        c_maxyd[tid] = max_y[tid] * ysd + ym;
    }
    __syncthreads();

    // ---- initial states (denormalized x_f_0) ----
    for (int idx = tid; idx < 16 * 32; idx += NTHREADS) {
        int m = idx >> 5, k = idx & 31;
        float v = x_f_0[(rbase + m) * NX + k] * c_xstd[k] + c_xmean[k];
        s_xf[idx] = v; s_xp[idx] = v;
        s_xb[idx] = x_b_0[(rbase + m) * NXB + k];
    }
    for (int idx = tid; idx < 16 * 8; idx += NTHREADS) {
        int m = idx >> 3, k = idx & 7;
        s_thf[idx] = theta_f_g[(rbase + m) * NTH + k];
        s_thh[idx] = theta_h_g[(rbase + m) * NTH + k];
    }
    __syncthreads();

    // ---- y0 = h([x_f0, th_h]) + h_b([x_f0, x_b0])  (unclipped) ----
    for (int idx = tid; idx < 16 * 64; idx += NTHREADS) {   // z_h
        int m = idx >> 6, k = idx & 63;
        float v = (k < 32) ? s_xp[m * 32 + k] : (k < 40) ? s_thh[m * 8 + (k - 32)] : 0.0f;
        s_z[m * ZS + k] = (half_t)v;
    }
    mlp_tile<2, 1>(s_z, ZS, ws + OFF_W1_H, b1_h, ws + OFF_W2_H, b2_h, NY,
                   s_hid, s_acc_y, false, tid);
    for (int idx = tid; idx < 16 * 64; idx += NTHREADS) {   // z_hb
        int m = idx >> 6, k = idx & 63;
        float v = (k < 32) ? s_xf[m * 32 + k] : s_xb[m * 32 + (k - 32)];
        s_z[m * ZS + k] = (half_t)v;
    }
    mlp_tile<2, 1>(s_z, ZS, ws + OFF_W1_HB, b1_hb, ws + OFF_W2_HB, b2_hb, NY,
                   s_hid, s_acc_y, true, tid);
    for (int idx = tid; idx < 16 * 8; idx += NTHREADS) {
        int m = idx >> 3, k = idx & 7;
        s_y[idx] = s_acc_y[m * 16 + k];
    }
    __syncthreads();

    // ======================= time scan =======================
    for (int t = 0; t < TSZ; ++t) {
        // ---- emit carry (pre-update) + load/denorm u_t ----
        for (int idx = tid; idx < 16 * 32; idx += NTHREADS) {
            int m = idx >> 5, k = idx & 31;
            int row = rbase + m;
            float xfn = (s_xf[idx] - c_xmean[k]) * c_xstdi[k];
            int o = (row * TSZ + t) * NX + k;
            out[OUT_XF + o] = xfn;
            out[OUT_XP + o] = xfn;       // reference returns x_f_norm for x_p
            out[OUT_XB + o] = s_xb[idx];
        }
        for (int idx = tid; idx < 16 * 8; idx += NTHREADS) {
            int m = idx >> 3, k = idx & 7;
            out[OUT_Y + ((rbase + m) * TSZ + t) * NY + k] =
                (s_y[idx] - c_ymean[k]) * c_ystdi[k];
        }
        for (int idx = tid; idx < 16 * 16; idx += NTHREADS) {
            int m = idx >> 4, k = idx & 15;
            s_ut[idx] = u[((rbase + m) * TSZ + t) * NU + k] * c_ustd[k] + c_umean[k];
        }
        __syncthreads();

        // ---- z_fb = [x_p, x_b, u_t] (pad to 96) ----
        for (int idx = tid; idx < 16 * ZS; idx += NTHREADS) {
            int m = idx / ZS, k = idx % ZS;
            float v = (k < 32) ? s_xp[m * 32 + k]
                    : (k < 64) ? s_xb[m * 32 + (k - 32)]
                    : (k < 80) ? s_ut[m * 16 + (k - 64)] : 0.0f;
            s_z[idx] = (half_t)v;
        }
        mlp_tile<3, 2>(s_z, ZS, ws + OFF_W1_GB, b1_gb, ws + OFF_W2_GB, b2_gb, NX,
                       s_hid, s_acc_gb, false, tid);
        mlp_tile<3, 2>(s_z, ZS, ws + OFF_W1_FB, b1_fb, ws + OFF_W2_FB, b2_fb, NX,
                       s_hid, s_acc_fb, false, tid);

        // ---- z_f = [x_p, u_t, theta_f] (pad to 64) ----
        for (int idx = tid; idx < 16 * 64; idx += NTHREADS) {
            int m = idx >> 6, k = idx & 63;
            float v = (k < 32) ? s_xp[m * 32 + k]
                    : (k < 48) ? s_ut[m * 16 + (k - 32)]
                    : (k < 56) ? s_thf[m * 8 + (k - 48)] : 0.0f;
            s_z[m * ZS + k] = (half_t)v;
        }
        mlp_tile<2, 2>(s_z, ZS, ws + OFF_W1_F, b1_f, ws + OFF_W2_F, b2_f, NX,
                       s_hid, s_acc_f, false, tid);

        // ---- state update (pre-clamp) ----
        for (int idx = tid; idx < 16 * 32; idx += NTHREADS) {
            float dxp = s_acc_f[idx];
            s_xf[idx] = s_xf[idx] + dxp + s_acc_fb[idx];
            s_xp[idx] = s_xp[idx] + dxp;
            s_xb[idx] = s_acc_gb[idx];
        }
        __syncthreads();

        // ---- y_n = h([x_p_n, th_h]) + h_b([x_f_n, x_b_n]) (pre-clamp states) ----
        for (int idx = tid; idx < 16 * 64; idx += NTHREADS) {
            int m = idx >> 6, k = idx & 63;
            float v = (k < 32) ? s_xp[m * 32 + k]
                    : (k < 40) ? s_thh[m * 8 + (k - 32)] : 0.0f;
            s_z[m * ZS + k] = (half_t)v;
        }
        mlp_tile<2, 1>(s_z, ZS, ws + OFF_W1_H, b1_h, ws + OFF_W2_H, b2_h, NY,
                       s_hid, s_acc_y, false, tid);
        for (int idx = tid; idx < 16 * 64; idx += NTHREADS) {
            int m = idx >> 6, k = idx & 63;
            float v = (k < 32) ? s_xf[m * 32 + k] : s_xb[m * 32 + (k - 32)];
            s_z[m * ZS + k] = (half_t)v;
        }
        mlp_tile<2, 1>(s_z, ZS, ws + OFF_W1_HB, b1_hb, ws + OFF_W2_HB, b2_hb, NY,
                       s_hid, s_acc_y, true, tid);

        // ---- clamps ----
        for (int idx = tid; idx < 16 * 32; idx += NTHREADS) {
            int k = idx & 31;
            s_xf[idx] = clipf(s_xf[idx], c_minxd[k], c_maxxd[k]);
            s_xp[idx] = clipf(s_xp[idx], c_minxd[k], c_maxxd[k]);
            s_xb[idx] = clipf(s_xb[idx], c_minxb[k], c_maxxb[k]);
        }
        for (int idx = tid; idx < 16 * 8; idx += NTHREADS) {
            int m = idx >> 3, k = idx & 7;
            s_y[idx] = clipf(s_acc_y[m * 16 + k], c_minyd[k], c_maxyd[k]);
        }
        __syncthreads();
    }
}

// ============================================================================
extern "C" void kernel_launch(void* const* d_in, const int* in_sizes, int n_in,
                              void* d_out, int out_size, void* d_ws, size_t ws_size,
                              hipStream_t stream) {
    (void)in_sizes; (void)n_in; (void)out_size; (void)ws_size;
    const float* x_f_0   = (const float*)d_in[0];
    const float* x_b_0   = (const float*)d_in[1];
    const float* u       = (const float*)d_in[2];
    const float* theta_f = (const float*)d_in[3];
    const float* theta_h = (const float*)d_in[4];
    const float* u_mean  = (const float*)d_in[5];
    const float* u_std   = (const float*)d_in[6];
    const float* x_mean  = (const float*)d_in[7];
    const float* x_std   = (const float*)d_in[8];
    const float* y_mean  = (const float*)d_in[9];
    const float* y_std   = (const float*)d_in[10];
    const float* min_x_b = (const float*)d_in[11];
    const float* max_x_b = (const float*)d_in[12];
    const float* min_x   = (const float*)d_in[13];
    const float* max_x   = (const float*)d_in[14];
    const float* min_y   = (const float*)d_in[15];
    const float* max_y   = (const float*)d_in[16];
    // f, f_b, g_b, h, h_b params: (W1, b1, W2, b2) each
    const float* W1_f  = (const float*)d_in[17]; const float* b1_f  = (const float*)d_in[18];
    const float* W2_f  = (const float*)d_in[19]; const float* b2_f  = (const float*)d_in[20];
    const float* W1_fb = (const float*)d_in[21]; const float* b1_fb = (const float*)d_in[22];
    const float* W2_fb = (const float*)d_in[23]; const float* b2_fb = (const float*)d_in[24];
    const float* W1_gb = (const float*)d_in[25]; const float* b1_gb = (const float*)d_in[26];
    const float* W2_gb = (const float*)d_in[27]; const float* b2_gb = (const float*)d_in[28];
    const float* W1_h  = (const float*)d_in[29]; const float* b1_h  = (const float*)d_in[30];
    const float* W2_h  = (const float*)d_in[31]; const float* b2_h  = (const float*)d_in[32];
    const float* W1_hb = (const float*)d_in[33]; const float* b1_hb = (const float*)d_in[34];
    const float* W2_hb = (const float*)d_in[35]; const float* b2_hb = (const float*)d_in[36];

    half_t* ws = (half_t*)d_ws;
    float* out = (float*)d_out;

    // ---- pack weights fp32 -> fp16 WMMA B-panels ----
    auto grid1 = [](int total) { return dim3((total + 255) / 256); };
    pack_w1<<<grid1(2 * 16384), 256, 0, stream>>>(W1_f,  56, ws + OFF_W1_F,  2);
    pack_w1<<<grid1(3 * 16384), 256, 0, stream>>>(W1_fb, 80, ws + OFF_W1_FB, 3);
    pack_w1<<<grid1(3 * 16384), 256, 0, stream>>>(W1_gb, 80, ws + OFF_W1_GB, 3);
    pack_w1<<<grid1(2 * 16384), 256, 0, stream>>>(W1_h,  40, ws + OFF_W1_H,  2);
    pack_w1<<<grid1(2 * 16384), 256, 0, stream>>>(W1_hb, 64, ws + OFF_W1_HB, 2);
    pack_w2<<<grid1(16 * 2 * 512), 256, 0, stream>>>(W2_f,  32, ws + OFF_W2_F,  2);
    pack_w2<<<grid1(16 * 2 * 512), 256, 0, stream>>>(W2_fb, 32, ws + OFF_W2_FB, 2);
    pack_w2<<<grid1(16 * 2 * 512), 256, 0, stream>>>(W2_gb, 32, ws + OFF_W2_GB, 2);
    pack_w2<<<grid1(16 * 1 * 512), 256, 0, stream>>>(W2_h,   8, ws + OFF_W2_H,  1);
    pack_w2<<<grid1(16 * 1 * 512), 256, 0, stream>>>(W2_hb,  8, ws + OFF_W2_HB, 1);

    // ---- persistent scan: 32 batch tiles of 16 rows ----
    sim_kernel<<<dim3(BSZ / 16), dim3(NTHREADS), 0, stream>>>(
        x_f_0, x_b_0, u, theta_f, theta_h,
        u_mean, u_std, x_mean, x_std, y_mean, y_std,
        min_x_b, max_x_b, min_x, max_x, min_y, max_y,
        b1_f, b2_f, b1_fb, b2_fb, b1_gb, b2_gb, b1_h, b2_h, b1_hb, b2_hb,
        ws, out);
}
